// SelfAttention_4466765987955
// MI455X (gfx1250) — compile-verified, matrix operand order checked by reading, not run on hardware
//
#include <hip/hip_runtime.h>
#include <hip/hip_bf16.h>

// ---------------------------------------------------------------------------
// Types
// ---------------------------------------------------------------------------
typedef __attribute__((ext_vector_type(16))) __bf16        v16bf;
typedef __attribute__((ext_vector_type(8)))  float         v8f;
typedef __attribute__((ext_vector_type(4)))  float         v4f;
typedef __attribute__((ext_vector_type(4)))  unsigned int  v4u;

union Frag {
    v16bf v;
    v4u   q[2];
};

#define DMODEL 768
#define SEQ    2048
#define NBATCH 8
#define MROWS  (NBATCH * SEQ)       // 16384
#define XPAD   (DMODEL + 8)         // 776: +16B row pad -> 4-bank rotation
#define SPAD   (SEQ + 8)            // 2056: same trick for the score block
#define WELEM  (DMODEL * DMODEL)    // 589824 elements per weight matrix

__device__ __forceinline__ unsigned short f2bf(float f) {
    __bf16 h = (__bf16)f;
    return __builtin_bit_cast(unsigned short, h);
}
__device__ __forceinline__ float bf2f(unsigned short u) {
    __bf16 h = __builtin_bit_cast(__bf16, u);
    return (float)h;
}
__device__ __forceinline__ unsigned int pk2(float lo, float hi) {
    return (unsigned int)f2bf(lo) | ((unsigned int)f2bf(hi) << 16);
}
__device__ __forceinline__ v4u pk8(const v4f a, const v4f b) {
    v4u r = { pk2(a[0], a[1]), pk2(a[2], a[3]),
              pk2(b[0], b[1]), pk2(b[2], b[3]) };
    return r;
}

__device__ __forceinline__ v8f wmma_bf16(v16bf a, v16bf b, v8f c) {
    return __builtin_amdgcn_wmma_f32_16x16x32_bf16(
        false, a, false, b, (short)0, c, false, false);
}

// ---------------------------------------------------------------------------
// Kernel 0: one-shot f32 -> bf16 conversion of Wq, Wk, Wv (L2-resident after).
// grid = 864 x 256, each thread packs 8 elements.
// ---------------------------------------------------------------------------
__global__ void __launch_bounds__(256)
cvtw_kernel(const float* __restrict__ Wq, const float* __restrict__ Wk,
            const float* __restrict__ Wv, unsigned short* __restrict__ o)
{
    const int t   = blockIdx.x * 256 + threadIdx.x;     // 0 .. 3*WELEM/8 - 1
    const int per = WELEM / 8;
    const int mat = t / per;
    const int r   = (t - mat * per) * 8;
    const float* src = (mat == 0) ? Wq : (mat == 1) ? Wk : Wv;
    v4f a = *(const v4f*)(src + r);
    v4f b = *(const v4f*)(src + r + 4);
    *(v4u*)(o + (size_t)mat * WELEM + r) = pk8(a, b);
}

// ---------------------------------------------------------------------------
// Kernel 1: fused QKV projection (all operands bf16 at the matrix pipe).
//   x tile (16x768) staged once per block into padded LDS (shared by 8 waves).
//   Weights read as pre-converted bf16 rows (contiguous b128 loads).
//   Q,K stored bf16 row-major; V stored bf16 transposed Vt[b][d][s].
// grid = (1024, 6), block = 256 (8 waves), wave w -> ntile = by*8 + w.
// ---------------------------------------------------------------------------
__global__ void __launch_bounds__(256)
qkv_kernel(const float* __restrict__ x,
           const unsigned short* __restrict__ Wqb,
           const unsigned short* __restrict__ Wkb,
           const unsigned short* __restrict__ Wvb,
           const float* __restrict__ bq, const float* __restrict__ bk,
           const float* __restrict__ bv,
           unsigned short* __restrict__ Qb,
           unsigned short* __restrict__ Kb,
           unsigned short* __restrict__ Vt)
{
    __shared__ __align__(16) unsigned short xs[16 * XPAD];

    const int wave  = threadIdx.x >> 5;
    const int lane  = threadIdx.x & 31;
    const int lm    = lane & 15;
    const int hi    = lane >> 4;
    const int mtile = blockIdx.x;
    const int ntile = blockIdx.y * 8 + wave;

    // ---- cooperative stage of the x tile: 12288 f32 -> bf16, 8 elems/thread
#pragma unroll
    for (int i = 0; i < 6; ++i) {
        const int e   = (threadIdx.x + i * 256) * 8;
        const int row = e / DMODEL;
        const int col = e - row * DMODEL;
        const float* p = x + (size_t)(mtile * 16 + row) * DMODEL + col;
        v4f a = *(const v4f*)(p);
        v4f b = *(const v4f*)(p + 4);
        *(v4u*)(&xs[row * XPAD + col]) = pk8(a, b);
    }
    __syncthreads();

    const unsigned short* xls  = &xs[lm * XPAD];
    const unsigned short* qrow = Wqb + (size_t)(ntile * 16 + lm) * DMODEL;
    const unsigned short* krow = Wkb + (size_t)(ntile * 16 + lm) * DMODEL;
    const unsigned short* vrow = Wvb + (size_t)(ntile * 16 + lm) * DMODEL;

    v8f accQ = {}, accK = {}, accV = {};

    // ---- software-pipelined K loop: prefetch k+32 while issuing 3 WMMAs
    int kb = hi * 8;
    v4u cx0 = *(const v4u*)(xls + kb),  cx1 = *(const v4u*)(xls + kb + 16);
    v4u cq0 = *(const v4u*)(qrow + kb), cq1 = *(const v4u*)(qrow + kb + 16);
    v4u ck0 = *(const v4u*)(krow + kb), ck1 = *(const v4u*)(krow + kb + 16);
    v4u cv0 = *(const v4u*)(vrow + kb), cv1 = *(const v4u*)(vrow + kb + 16);

#pragma unroll
    for (int k0 = 0; k0 < DMODEL; k0 += 32) {
        v4u nx0 = cx0, nx1 = cx1, nq0 = cq0, nq1 = cq1;
        v4u nk0 = ck0, nk1 = ck1, nv0 = cv0, nv1 = cv1;
        if (k0 + 32 < DMODEL) {
            const int kn = k0 + 32 + hi * 8;
            nx0 = *(const v4u*)(xls + kn);  nx1 = *(const v4u*)(xls + kn + 16);
            nq0 = *(const v4u*)(qrow + kn); nq1 = *(const v4u*)(qrow + kn + 16);
            nk0 = *(const v4u*)(krow + kn); nk1 = *(const v4u*)(krow + kn + 16);
            nv0 = *(const v4u*)(vrow + kn); nv1 = *(const v4u*)(vrow + kn + 16);
        }
        Frag xf, qf, kf, vf;
        xf.q[0] = cx0; xf.q[1] = cx1;
        qf.q[0] = cq0; qf.q[1] = cq1;
        kf.q[0] = ck0; kf.q[1] = ck1;
        vf.q[0] = cv0; vf.q[1] = cv1;
        accQ = wmma_bf16(qf.v, xf.v, accQ);   // D[n][m] = (x Wq^T)^T tile
        accK = wmma_bf16(kf.v, xf.v, accK);
        accV = wmma_bf16(xf.v, vf.v, accV);   // D[m][n] = x Wv^T tile
        cx0 = nx0; cx1 = nx1; cq0 = nq0; cq1 = nq1;
        ck0 = nk0; ck1 = nk1; cv0 = nv0; cv1 = nv1;
    }

    // ---- Q / K: lane -> m = lm, vgpr r -> n = r + 8*hi: packed b128 store
    {
        const int mglob = mtile * 16 + lm;
        const int nbase = ntile * 16 + 8 * hi;
        v4f bq0 = *(const v4f*)(bq + nbase), bq1 = *(const v4f*)(bq + nbase + 4);
        v4f bk0 = *(const v4f*)(bk + nbase), bk1 = *(const v4f*)(bk + nbase + 4);
        float qv[8], kv[8];
#pragma unroll
        for (int r = 0; r < 8; ++r) {
            qv[r] = accQ[r] + ((r < 4) ? bq0[r] : bq1[r - 4]);
            kv[r] = accK[r] + ((r < 4) ? bk0[r] : bk1[r - 4]);
        }
        v4u qpk = { pk2(qv[0], qv[1]), pk2(qv[2], qv[3]),
                    pk2(qv[4], qv[5]), pk2(qv[6], qv[7]) };
        v4u kpk = { pk2(kv[0], kv[1]), pk2(kv[2], kv[3]),
                    pk2(kv[4], kv[5]), pk2(kv[6], kv[7]) };
        *(v4u*)(Qb + (size_t)mglob * DMODEL + nbase) = qpk;
        *(v4u*)(Kb + (size_t)mglob * DMODEL + nbase) = kpk;
    }

    // ---- V transposed: lane -> d = ntile*16+lm, vgpr r -> s consecutive
    {
        const int b     = mtile >> 7;
        const int sbase = (mtile & 127) * 16 + 8 * hi;
        const int dglob = ntile * 16 + lm;
        const float bias = bv[dglob];
        float vv[8];
#pragma unroll
        for (int r = 0; r < 8; ++r) vv[r] = accV[r] + bias;
        v4u vpk = { pk2(vv[0], vv[1]), pk2(vv[2], vv[3]),
                    pk2(vv[4], vv[5]), pk2(vv[6], vv[7]) };
        *(v4u*)(Vt + (size_t)b * DMODEL * SEQ + (size_t)dglob * SEQ + sbase) = vpk;
    }
}

// ---------------------------------------------------------------------------
// Kernel 2: attention for one (batch, 16-query-row tile).
//   LDS: padded Q tile (staged once) + padded 16x2048 bf16 score block.
// grid = (128, 8), block = 256.
// ---------------------------------------------------------------------------
__global__ void __launch_bounds__(256)
attn_kernel(const unsigned short* __restrict__ Qb,
            const unsigned short* __restrict__ Kb,
            const unsigned short* __restrict__ Vt,
            float* __restrict__ out)
{
    __shared__ __align__(16) unsigned short sP[16 * SPAD];  // scores, padded
    __shared__ __align__(16) unsigned short qs[16 * XPAD];  // Q tile, padded
    __shared__ float row_inv[16];

    const int wave  = threadIdx.x >> 5;
    const int lane  = threadIdx.x & 31;
    const int lm    = lane & 15;
    const int hi    = lane >> 4;
    const int qtile = blockIdx.x;
    const int b     = blockIdx.y;

    const unsigned short* Qt  = Qb + ((size_t)(b * SEQ + qtile * 16)) * DMODEL;
    const unsigned short* Kbb = Kb + (size_t)b * SEQ * DMODEL;
    const unsigned short* Vtb = Vt + (size_t)b * DMODEL * SEQ;
    const float scale = 0.03608439182435161f;   // 1/sqrt(768)

    // ---- stage Q tile into padded LDS (bf16 already), 8 elems/thread
#pragma unroll
    for (int i = 0; i < 6; ++i) {
        const int e   = (threadIdx.x + i * 256) * 8;
        const int row = e / DMODEL;
        const int col = e - row * DMODEL;
        *(v4u*)(&qs[row * XPAD + col]) =
            *(const v4u*)(Qt + (size_t)row * DMODEL + col);
    }
    __syncthreads();

    // ---------------- Pass A: S = K Q^T * scale -> LDS ----------------
    const unsigned short* qls = &qs[lm * XPAD];
    for (int j = wave; j < SEQ / 16; j += 8) {
        const unsigned short* krow = Kbb + (size_t)(j * 16 + lm) * DMODEL;
        if (j + 8 < SEQ / 16)   // prefetch next key tile row for this lane
            __builtin_prefetch(Kbb + (size_t)((j + 8) * 16 + lm) * DMODEL, 0, 0);
        v8f acc = {};
        int kb = hi * 8;
        v4u ck0 = *(const v4u*)(krow + kb), ck1 = *(const v4u*)(krow + kb + 16);
#pragma unroll
        for (int k0 = 0; k0 < DMODEL; k0 += 32) {
            v4u nk0 = ck0, nk1 = ck1;
            if (k0 + 32 < DMODEL) {
                const int kn = k0 + 32 + hi * 8;
                nk0 = *(const v4u*)(krow + kn);
                nk1 = *(const v4u*)(krow + kn + 16);
            }
            const int kq = k0 + hi * 8;
            Frag qf, kf;
            qf.q[0] = *(const v4u*)(qls + kq);
            qf.q[1] = *(const v4u*)(qls + kq + 16);
            kf.q[0] = ck0; kf.q[1] = ck1;
            acc = wmma_bf16(kf.v, qf.v, acc);   // D[n][m]: rows land contiguous
            ck0 = nk0; ck1 = nk1;
        }
        v4u spk = { pk2(acc[0] * scale, acc[1] * scale),
                    pk2(acc[2] * scale, acc[3] * scale),
                    pk2(acc[4] * scale, acc[5] * scale),
                    pk2(acc[6] * scale, acc[7] * scale) };
        *(v4u*)(&sP[lm * SPAD + j * 16 + 8 * hi]) = spk;
    }
    __syncthreads();

    // ---------------- Pass B: rowwise softmax ----------------
    {
        const int r = 2 * wave + hi;             // 16 lanes per row
        unsigned short* srow = &sP[r * SPAD];
        float mx = -__builtin_inff();
#pragma unroll 8
        for (int i = 0; i < SEQ / 16; ++i)
            mx = fmaxf(mx, bf2f(srow[lm + 16 * i]));
#pragma unroll
        for (int m = 1; m <= 8; m <<= 1)
            mx = fmaxf(mx, __shfl_xor(mx, m, 32));
        float sum = 0.0f;
#pragma unroll 8
        for (int i = 0; i < SEQ / 16; ++i) {
            const int idx = lm + 16 * i;
            float e = __expf(bf2f(srow[idx]) - mx);
            srow[idx] = f2bf(e);                 // unnormalized p, in place
            sum += e;
        }
#pragma unroll
        for (int m = 1; m <= 8; m <<= 1)
            sum += __shfl_xor(sum, m, 32);
        if (lm == 0) row_inv[r] = 1.0f / sum;
    }
    __syncthreads();

    // ---------------- Pass C: out = (P @ V) * row_inv ----------------
    float* orow = out + ((size_t)(b * SEQ + qtile * 16)) * DMODEL;
    const unsigned short* pls = &sP[lm * SPAD];
    for (int nt = wave; nt < DMODEL / 16; nt += 8) {
        const unsigned short* vrow = Vtb + (size_t)(nt * 16 + lm) * SEQ;
        v8f acc = {};
        int kb = hi * 8;
        v4u cv0 = *(const v4u*)(vrow + kb), cv1 = *(const v4u*)(vrow + kb + 16);
#pragma unroll 8
        for (int k0 = 0; k0 < SEQ; k0 += 32) {
            v4u nv0 = cv0, nv1 = cv1;
            if (k0 + 32 < SEQ) {
                const int kn = k0 + 32 + hi * 8;
                nv0 = *(const v4u*)(vrow + kn);
                nv1 = *(const v4u*)(vrow + kn + 16);
            }
            const int kp = k0 + hi * 8;
            Frag pf, vf;
            pf.q[0] = *(const v4u*)(pls + kp);
            pf.q[1] = *(const v4u*)(pls + kp + 16);
            vf.q[0] = cv0; vf.q[1] = cv1;
            acc = wmma_bf16(pf.v, vf.v, acc);    // D[m][n]
            cv0 = nv0; cv1 = nv1;
        }
#pragma unroll
        for (int r = 0; r < 8; ++r) {
            const int m = r + 8 * hi;
            orow[(size_t)m * DMODEL + nt * 16 + lm] = acc[r] * row_inv[m];
        }
    }
}

// ---------------------------------------------------------------------------
// Launcher
// ---------------------------------------------------------------------------
extern "C" void kernel_launch(void* const* d_in, const int* in_sizes, int n_in,
                              void* d_out, int out_size, void* d_ws, size_t ws_size,
                              hipStream_t stream) {
    const float* x  = (const float*)d_in[0];
    const float* Wk = (const float*)d_in[1];
    const float* bk = (const float*)d_in[2];
    const float* Wq = (const float*)d_in[3];
    const float* bq = (const float*)d_in[4];
    const float* Wv = (const float*)d_in[5];
    const float* bv = (const float*)d_in[6];
    float* out = (float*)d_out;

    // workspace layout (bf16 elements)
    unsigned short* Qb  = (unsigned short*)d_ws;
    unsigned short* Kb  = Qb  + (size_t)MROWS * DMODEL;
    unsigned short* Vt  = Kb  + (size_t)MROWS * DMODEL;
    unsigned short* Wqb = Vt  + (size_t)MROWS * DMODEL;
    unsigned short* Wkb = Wqb + (size_t)WELEM;
    unsigned short* Wvb = Wkb + (size_t)WELEM;

    cvtw_kernel<<<dim3(3 * WELEM / 8 / 256), 256, 0, stream>>>(Wq, Wk, Wv, Wqb);

    dim3 g1(MROWS / 16, (DMODEL / 16) / 8, 1);   // (1024, 6)
    qkv_kernel<<<g1, 256, 0, stream>>>(x, Wqb, Wkb, Wvb, bq, bk, bv, Qb, Kb, Vt);

    dim3 g2(SEQ / 16, NBATCH, 1);                // (128, 8)
    attn_kernel<<<g2, 256, 0, stream>>>(Qb, Kb, Vt, out);
}
